// RegionProposalNetwork_10015863734528
// MI455X (gfx1250) — compile-verified, matrix operand order checked by reading, not run on hardware
//
#include <hip/hip_runtime.h>
#include <hip/hip_bf16.h>
#include <stdint.h>

// ---------------------------------------------------------------------------
// Types for CDNA5 WMMA
// ---------------------------------------------------------------------------
typedef __bf16        v16bf __attribute__((ext_vector_type(16)));
typedef float         v8f   __attribute__((ext_vector_type(8)));
typedef unsigned int  v4u   __attribute__((ext_vector_type(4)));

#define B_      16
#define CIN     2048
#define HW      256           // 16x16
#define HP      18            // padded spatial dim
#define HIDDEN_ 512
#define KTOT    (CIN * 9)     // 18432

union FragAB {
    v16bf v;
    v4u   q[2];
};

// ---------------------------------------------------------------------------
// f32 -> bf16 (round-to-nearest-even)
// ---------------------------------------------------------------------------
__device__ __forceinline__ unsigned short f2bf(float x) {
    unsigned u = __float_as_uint(x);
    u += 0x7FFFu + ((u >> 16) & 1u);
    return (unsigned short)(u >> 16);
}

// Weights: [oc][ic][3][3] f32  ->  [oc][khw][ic] bf16  (K = khw*2048 + ic)
__global__ void cvt_w_kernel(const float* __restrict__ src,
                             unsigned short* __restrict__ dst) {
    int tid = blockIdx.x * blockDim.x + threadIdx.x;       // oc*9*2048 + khw*2048 + ic
    if (tid >= HIDDEN_ * 9 * CIN) return;
    int oc  = tid / (9 * CIN);
    int rem = tid - oc * (9 * CIN);
    int khw = rem / CIN;
    int ic  = rem - khw * CIN;
    dst[tid] = f2bf(src[(oc * CIN + ic) * 9 + khw]);
}

// Feature map: [16][2048][16][16] f32 -> zero-padded NHWC bf16 [16][18][18][2048]
__global__ void cvt_x_pad_kernel(const float* __restrict__ src,
                                 unsigned short* __restrict__ dst) {
    int tid = blockIdx.x * blockDim.x + threadIdx.x;       // ((b*18+hh)*18+ww)*2048 + ic
    if (tid >= B_ * HP * HP * CIN) return;
    int ic  = tid & (CIN - 1);
    int sp  = tid >> 11;                                   // (b*18+hh)*18+ww
    int ww  = sp % HP;
    int t2  = sp / HP;
    int hh  = t2 % HP;
    int b   = t2 / HP;
    unsigned short val = 0;
    if (hh >= 1 && hh <= 16 && ww >= 1 && ww <= 16)
        val = f2bf(src[((b * CIN + ic) * 16 + (hh - 1)) * 16 + (ww - 1)]);
    dst[tid] = val;
}

// ---------------------------------------------------------------------------
// Implicit-GEMM 3x3 conv via v_wmma_f32_16x16x32_bf16
//   M = oc (512), N = pixel (256/image), K = 9*2048 (khw-major)
//   grid = (256 ntiles, 4), block = 256 (8 waves); one 16x16 tile per wave.
//   Hot loop: 2x b128 (A) + 2x b128 (B) + 1x WMMA, no branches/divides.
// ---------------------------------------------------------------------------
__global__ __launch_bounds__(256)
void conv3x3_wmma_kernel(const unsigned short* __restrict__ xp,   // padded NHWC bf16
                         const unsigned short* __restrict__ wb,   // [oc][khw][ic] bf16
                         const float* __restrict__ b1,
                         float* __restrict__ convout) {           // [16,512,256] f32 post-ReLU
    const int lane  = threadIdx.x & 31;
    const int wave  = threadIdx.x >> 5;
    const int mtile = blockIdx.y * 8 + wave;   // 0..31
    const int ntile = blockIdx.x;              // 0..255
    const int bimg  = ntile >> 4;
    const int nt    = ntile & 15;
    const unsigned half = (unsigned)(lane >> 4);
    const int      l15  = lane & 15;

    const int oc0 = mtile * 16;
    // A row for this lane (lane 0..15 and 16..31 both map to rows M=0..15;
    // `half` selects the K sub-segment per the 16-bit A VGPR layout).
    const unsigned short* wrow = wb + (size_t)(oc0 + l15) * (9 * CIN);
    // B column = output pixel
    const int p = nt * 16 + l15;
    const int h = p >> 4, w = p & 15;

    v8f c = {0.f, 0.f, 0.f, 0.f, 0.f, 0.f, 0.f, 0.f};

#pragma unroll
    for (int khw = 0; khw < 9; ++khw) {
        const int kh = khw / 3, kw = khw - 3 * (khw / 3);
        const unsigned short* arow = wrow + khw * CIN;
        const unsigned short* brow =
            xp + ((size_t)((bimg * HP + h + kh) * HP + (w + kw))) * CIN;
#pragma unroll 4
        for (int kc = 0; kc < CIN / 32; ++kc) {
            const unsigned k0 = (unsigned)kc * 32u + half * 8u;
            FragAB a, bm;
            a.q[0]  = *(const v4u*)(arow + k0);
            a.q[1]  = *(const v4u*)(arow + k0 + 16);
            bm.q[0] = *(const v4u*)(brow + k0);
            bm.q[1] = *(const v4u*)(brow + k0 + 16);
            c = __builtin_amdgcn_wmma_f32_16x16x32_bf16(
                    /*neg_a=*/false, a.v, /*neg_b=*/false, bm.v,
                    /*c_mod=*/(short)0, c, /*reuse_a=*/false, /*reuse_b=*/false);
        }
    }

    // C/D layout: VGPR v -> M = v + 8*half, N = l15. Fuse bias + ReLU.
#pragma unroll
    for (int v = 0; v < 8; ++v) {
        const int oc = oc0 + v + 8 * (int)half;
        float val = c[v] + b1[oc];
        val = val > 0.f ? val : 0.f;
        convout[((unsigned)bimg * HIDDEN_ + (unsigned)oc) * HW + (unsigned)p] = val;
    }
}

// ---------------------------------------------------------------------------
// 1x1 heads: out[b,o,p] = sum_c convout[b,c,p] * Wh[o,c] + bh[o]
// ---------------------------------------------------------------------------
__global__ void head_gemm_kernel(const float* __restrict__ convout,
                                 const float* __restrict__ Wh,
                                 const float* __restrict__ bh,
                                 float* __restrict__ outv, int OCH) {
    int tid = blockIdx.x * blockDim.x + threadIdx.x;
    int total = B_ * OCH * HW;
    if (tid >= total) return;
    int b   = tid / (OCH * HW);
    int rem = tid - b * (OCH * HW);
    int o   = rem / HW;
    int p   = rem - o * HW;
    const float* xpR = convout + (unsigned)b * HIDDEN_ * HW + p;
    const float* wp  = Wh + o * HIDDEN_;
    float acc = bh[o];
#pragma unroll 4
    for (int cc = 0; cc < HIDDEN_; ++cc)
        acc = fmaf(xpR[cc * HW], wp[cc], acc);
    outv[tid] = acc;
}

// ---------------------------------------------------------------------------
// Loss kernels (deterministic two-stage reduction, LDS tree)
// ---------------------------------------------------------------------------
__device__ __forceinline__ float softplus_(float x) {
    return fmaxf(x, 0.f) + log1pf(expf(-fabsf(x)));
}

__global__ __launch_bounds__(256)
void pos_loss_kernel(const float* __restrict__ conf, const float* __restrict__ reg,
                     const int* __restrict__ pos_ind,
                     const float* __restrict__ anc, const float* __restrict__ gt,
                     float* __restrict__ out_prop,   // d_out + 1, [2048,4]
                     float* __restrict__ posCls, float* __restrict__ posReg) {
    __shared__ float s1[256], s2[256];
    const int i   = blockIdx.x * 256 + threadIdx.x;    // 0..2047
    const int idx = pos_ind[i];

    const float cp  = conf[idx];
    const float cls = softplus_(-cp);                  // BCE(logit, target=1)

    const float off0 = reg[4 * idx + 0], off1 = reg[4 * idx + 1];
    const float off2 = reg[4 * idx + 2], off3 = reg[4 * idx + 3];

    const float ax1 = anc[4 * i + 0], ay1 = anc[4 * i + 1];
    const float ax2 = anc[4 * i + 2], ay2 = anc[4 * i + 3];
    const float acx = 0.5f * (ax1 + ax2), acy = 0.5f * (ay1 + ay2);
    const float aw  = ax2 - ax1,          ah  = ay2 - ay1;

    // proposals
    const float pcx = acx + off0 * aw, pcy = acy + off1 * ah;
    const float pw  = aw * expf(off2), ph  = ah * expf(off3);
    out_prop[4 * i + 0] = pcx - 0.5f * pw;
    out_prop[4 * i + 1] = pcy - 0.5f * ph;
    out_prop[4 * i + 2] = pcx + 0.5f * pw;
    out_prop[4 * i + 3] = pcy + 0.5f * ph;

    // gt offsets + smooth-L1 (beta=1)
    const float gx1 = gt[4 * i + 0], gy1 = gt[4 * i + 1];
    const float gx2 = gt[4 * i + 2], gy2 = gt[4 * i + 3];
    const float gcx = 0.5f * (gx1 + gx2), gcy = 0.5f * (gy1 + gy2);
    const float gw  = gx2 - gx1,          gh  = gy2 - gy1;
    const float t0 = (gcx - acx) / aw, t1 = (gcy - acy) / ah;
    const float t2 = logf(gw / aw),    t3 = logf(gh / ah);

    float dv[4] = {off0 - t0, off1 - t1, off2 - t2, off3 - t3};
    float regsum = 0.f;
#pragma unroll
    for (int j = 0; j < 4; ++j) {
        const float ad = fabsf(dv[j]);
        regsum += (ad < 1.f) ? 0.5f * dv[j] * dv[j] : ad - 0.5f;
    }

    s1[threadIdx.x] = cls;
    s2[threadIdx.x] = regsum;
    __syncthreads();
    for (int st = 128; st > 0; st >>= 1) {
        if (threadIdx.x < st) {
            s1[threadIdx.x] += s1[threadIdx.x + st];
            s2[threadIdx.x] += s2[threadIdx.x + st];
        }
        __syncthreads();
    }
    if (threadIdx.x == 0) {
        posCls[blockIdx.x] = s1[0];
        posReg[blockIdx.x] = s2[0];
    }
}

__global__ __launch_bounds__(256)
void neg_loss_kernel(const float* __restrict__ conf, const int* __restrict__ neg_ind,
                     float* __restrict__ negCls) {
    __shared__ float s1[256];
    const int i = blockIdx.x * 256 + threadIdx.x;      // 0..2047
    const float cn = conf[neg_ind[i]];
    s1[threadIdx.x] = softplus_(cn);                   // BCE(logit, target=0)
    __syncthreads();
    for (int st = 128; st > 0; st >>= 1) {
        if (threadIdx.x < st) s1[threadIdx.x] += s1[threadIdx.x + st];
        __syncthreads();
    }
    if (threadIdx.x == 0) negCls[blockIdx.x] = s1[0];
}

__global__ void finalize_kernel(const float* __restrict__ posCls,
                                const float* __restrict__ posReg,
                                const float* __restrict__ negCls,
                                float* __restrict__ out0) {
    if (threadIdx.x == 0 && blockIdx.x == 0) {
        float pc = 0.f, pr = 0.f, nc = 0.f;
        for (int i = 0; i < 8; ++i) { pc += posCls[i]; pr += posReg[i]; nc += negCls[i]; }
        out0[0] = (pc + nc) / 16.f + 5.f * (pr / 16.f);
    }
}

// ---------------------------------------------------------------------------
// Launcher
// ---------------------------------------------------------------------------
extern "C" void kernel_launch(void* const* d_in, const int* in_sizes, int n_in,
                              void* d_out, int out_size, void* d_ws, size_t ws_size,
                              hipStream_t stream) {
    const float* feature_map = (const float*)d_in[0];   // [16,2048,16,16]
    const float* pos_coords  = (const float*)d_in[1];   // [2048,4]
    const float* gt_pos      = (const float*)d_in[2];   // [2048,4]
    const float* W1          = (const float*)d_in[3];   // [512,2048,3,3]
    const float* b1          = (const float*)d_in[4];   // [512]
    const float* Wc          = (const float*)d_in[5];   // [9,512]
    const float* bc          = (const float*)d_in[6];   // [9]
    const float* Wr          = (const float*)d_in[7];   // [36,512]
    const float* br          = (const float*)d_in[8];   // [36]
    const int*   pos_ind     = (const int*)d_in[9];     // [2048]
    const int*   neg_ind     = (const int*)d_in[10];    // [2048]
    float*       out         = (float*)d_out;           // [1 + 2048*4]

    uint8_t* ws = (uint8_t*)d_ws;
    const size_t N_XP = (size_t)B_ * HP * HP * CIN;     // 10,616,832 (padded NHWC)
    const size_t N_W  = (size_t)HIDDEN_ * KTOT;         // 9,437,184
    size_t off = 0;
    unsigned short* xp = (unsigned short*)(ws + off); off += N_XP * 2;           // ~21 MB
    unsigned short* wb = (unsigned short*)(ws + off); off += N_W * 2;            // ~18 MB
    float* convout = (float*)(ws + off); off += (size_t)B_ * HIDDEN_ * HW * 4;   //  8 MB
    float* conf    = (float*)(ws + off); off += (size_t)B_ * 9  * HW * 4;
    float* reg     = (float*)(ws + off); off += (size_t)B_ * 36 * HW * 4;
    float* posCls  = (float*)(ws + off); off += 8 * 4;
    float* posReg  = (float*)(ws + off); off += 8 * 4;
    float* negCls  = (float*)(ws + off); off += 8 * 4;

    // 1) layout transforms + f32 -> bf16
    cvt_x_pad_kernel<<<(int)((N_XP + 255) / 256), 256, 0, stream>>>(feature_map, xp);
    cvt_w_kernel<<<(int)((N_W + 255) / 256), 256, 0, stream>>>(W1, wb);

    // 2) conv3x3 + bias + ReLU via WMMA bf16 (pure b128-load + wmma hot loop)
    conv3x3_wmma_kernel<<<dim3(256, 4), 256, 0, stream>>>(xp, wb, b1, convout);

    // 3) 1x1 heads
    head_gemm_kernel<<<(B_ * 9  * HW + 255) / 256, 256, 0, stream>>>(convout, Wc, bc, conf, 9);
    head_gemm_kernel<<<(B_ * 36 * HW + 255) / 256, 256, 0, stream>>>(convout, Wr, br, reg, 36);

    // 4) gathers + proposals + losses (deterministic reductions)
    pos_loss_kernel<<<8, 256, 0, stream>>>(conf, reg, pos_ind, pos_coords, gt_pos,
                                           out + 1, posCls, posReg);
    neg_loss_kernel<<<8, 256, 0, stream>>>(conf, neg_ind, negCls);
    finalize_kernel<<<1, 32, 0, stream>>>(posCls, posReg, negCls, out);
}
